// LSTMDecoder_90280212562570
// MI455X (gfx1250) — matrix operand order repaired and checked by reading, not compiled, not run on hardware
//
#include <hip/hip_runtime.h>
#include <hip/hip_bf16.h>
#include <math.h>

// ---------------------------------------------------------------------------
// LSTM decoder for MI455X (gfx1250, wave32, WMMA).
//
// Workspace layout (needs ~140 MB; harness preallocates d_ws):
//   [0]                whh_aug : bf16 [2][2048][544]   (W_hh | folded input W) = 4,456,448 B
//   [+4456448]         ctxterm : f32  [2][64][2048]    (ctx@W2^T + biases)     = 1,048,576 B
//   [+5505024]         hseq    : bf16 [2][64][1024][512]                       = 134,217,728 B
// ---------------------------------------------------------------------------

typedef __attribute__((ext_vector_type(16))) __bf16 v16bf;
typedef __attribute__((ext_vector_type(8)))  __bf16 v8bf;
typedef __attribute__((ext_vector_type(8)))  float  v8f;

#define T_STEPS 1024
#define H       512
#define KAUG    544            // 512 h + 10/6 folded input + pad -> 17 k-tiles of 32
#define NKT     17
#define LDS_STRIDE 560         // halfs per row (544 + 16 pad), 16B-aligned tiles

// Fast transcendentals: keep the serial elementwise path on the TRANS pipe
// (v_exp_f32 / v_rcp_f32 / v_tanh_f32) instead of IEEE v_div_* sequences.
__device__ __forceinline__ float fast_sigmoid(float x) {
  return __builtin_amdgcn_rcpf(1.f + __expf(-x));
}
__device__ __forceinline__ float fast_tanh(float x) {
#if __has_builtin(__builtin_amdgcn_tanhf)
  return __builtin_amdgcn_tanhf(x);
#elif __has_builtin(__builtin_amdgcn_tanh_f32)
  return __builtin_amdgcn_tanh_f32(x);
#else
  // tanh(x) = 1 - 2/(exp(2x)+1)
  return 1.f - 2.f * __builtin_amdgcn_rcpf(1.f + __expf(2.f * x));
#endif
}

// ---------------------------------------------------------------------------
// Kernel 1: build augmented bf16 B-matrix rows: [W_hh[g,0:512] | Wfold[g,0:KIN] | 0]
//   Wfold[g,kin] = sum_d W_ih[g,d] * W_emb[d,kin]   (embedding folded into gates)
// grid (2048, 2), block 256
// ---------------------------------------------------------------------------
__global__ void prep_w_kernel(const float* __restrict__ whh_cmd,
                              const float* __restrict__ wih_cmd,
                              const float* __restrict__ wemb_cmd,
                              const float* __restrict__ whh_crd,
                              const float* __restrict__ wih_crd,
                              const float* __restrict__ wemb_crd,
                              __bf16* __restrict__ whh_aug) {
  const int g = blockIdx.x;
  const int lstm = blockIdx.y;
  const float* whh  = lstm ? whh_crd  : whh_cmd;
  const float* wih  = lstm ? wih_crd  : wih_cmd;
  const float* wemb = lstm ? wemb_crd : wemb_cmd;
  const int KIN = lstm ? 6 : 10;
  __bf16* out = whh_aug + ((size_t)lstm * 2048 + g) * KAUG;

  for (int k = threadIdx.x; k < H; k += 256)
    out[k] = (__bf16)whh[(size_t)g * H + k];

  for (int k = H + (int)threadIdx.x; k < KAUG; k += 256) {
    const int kin = k - H;
    float s = 0.f;
    if (kin < KIN) {
      const float* wr = wih + (size_t)g * 768;
      for (int d = 0; d < H; ++d) s += wr[d] * wemb[(size_t)d * KIN + kin];
    }
    out[k] = (__bf16)s;
  }
}

// ---------------------------------------------------------------------------
// Kernel 2: per-(lstm,b) constant gate bias (accumulator init, constant over T):
//   ctxterm[l][b][g] = b_ih[g]+b_hh[g] + sum_d b_emb[d]*W_ih[g,d]
//                    + sum_k ctx[b,k]*W_ih[g,512+k]
// grid (8, 64, 2), block 256
// ---------------------------------------------------------------------------
__global__ void prep_ctx_kernel(const float* __restrict__ context,
                                const float* __restrict__ wih_cmd,
                                const float* __restrict__ bemb_cmd,
                                const float* __restrict__ bih_cmd,
                                const float* __restrict__ bhh_cmd,
                                const float* __restrict__ wih_crd,
                                const float* __restrict__ bemb_crd,
                                const float* __restrict__ bih_crd,
                                const float* __restrict__ bhh_crd,
                                float* __restrict__ ctxterm) {
  const int g = blockIdx.x * 256 + threadIdx.x;
  const int b = blockIdx.y;
  const int lstm = blockIdx.z;
  const float* wih  = lstm ? wih_crd  : wih_cmd;
  const float* bemb = lstm ? bemb_crd : bemb_cmd;
  float s = lstm ? (bih_crd[g] + bhh_crd[g]) : (bih_cmd[g] + bhh_cmd[g]);
  const float* wr = wih + (size_t)g * 768;
  for (int d = 0; d < H; ++d) s += bemb[d] * wr[d];
  const float* cx = context + (size_t)b * 256;
  for (int k = 0; k < 256; ++k) s += cx[k] * wr[H + k];
  ctxterm[((size_t)lstm * 64 + b) * 2048 + g] = s;
}

// ---------------------------------------------------------------------------
// Kernel 3: the recurrence. grid (4 batch-groups, 2 lstms) x 512 threads
// (16 waves). Wave w owns hidden columns [w*32, w*32+32): its 8 WMMA tiles
// cover gates i/f/g/o for that slice, so c/h updates are wave-local.
// Per step: gates = [h | x_fold] @ whh_aug^T  (17 k-tiles x 8 N-tiles WMMA),
// accumulators seeded from ctxterm registers. h_t kept in LDS (bf16),
// h_seq streamed to workspace.
// ---------------------------------------------------------------------------
__global__ void __launch_bounds__(512)
lstm_rec_kernel(const float* __restrict__ x,          // [64,1024,16]
                const float* __restrict__ ctxterm,    // [2][64][2048]
                const __bf16* __restrict__ whh_aug,   // [2][2048][544]
                __bf16* __restrict__ hseq) {          // [2][64][1024][512]
  const int lstm = blockIdx.y;
  const int bg   = blockIdx.x;         // batch group: rows bg*16 .. bg*16+15
  const int tid  = threadIdx.x;
  const int wv   = tid >> 5;           // wave 0..15
  const int l    = tid & 31;           // lane
  const int mh   = l >> 4;             // lane half (0/1)
  const int nl   = l & 15;             // N (and A's M) within tile

  __shared__ __bf16 hbuf[16 * LDS_STRIDE];

  for (int i = tid; i < 16 * LDS_STRIDE; i += 512) hbuf[i] = (__bf16)0.f;

  const __bf16* wB = whh_aug + (size_t)lstm * 2048 * KAUG;
  const float*  ct = ctxterm + ((size_t)lstm * 64 + bg * 16) * 2048;
  const int KIN  = lstm ? 6 : 10;
  const int XOFF = lstm ? 10 : 0;

  // Accumulator init values (constant over t): C/D layout m = r + 8*mh, n = nl.
  v8f cinit[4][2];
#pragma unroll
  for (int q = 0; q < 4; ++q)
#pragma unroll
    for (int j = 0; j < 2; ++j) {
      const int n = q * H + wv * 32 + j * 16 + nl;
#pragma unroll
      for (int r = 0; r < 8; ++r)
        cinit[q][j][r] = ct[(size_t)(r + 8 * mh) * 2048 + n];
    }

  v8f cst[2];
#pragma unroll
  for (int j = 0; j < 2; ++j)
#pragma unroll
    for (int r = 0; r < 8; ++r) cst[j][r] = 0.f;

  __syncthreads();
  // folded input for t=0 into LDS cols [512, 512+KIN)
  if (tid < 256) {
    const int m = tid >> 4, k = tid & 15;
    if (k < KIN) {
      float v = x[(((size_t)(bg * 16 + m)) * T_STEPS + 0) * 16 + XOFF + k];
      hbuf[m * LDS_STRIDE + H + k] = (__bf16)v;
    }
  }
  __syncthreads();

  for (int t = 0; t < T_STEPS; ++t) {
    v8f acc[4][2];
#pragma unroll
    for (int q = 0; q < 4; ++q)
#pragma unroll
      for (int j = 0; j < 2; ++j) acc[q][j] = cinit[q][j];

#pragma unroll 1
    for (int kt = 0; kt < NKT; ++kt) {
      // A tile (16x32 bf16): lane m = nl; halves {0..15}->K(kt*32 + mh*16 + 0..15),
      // contiguous, matching the B fill.
      const __bf16* pa = &hbuf[nl * LDS_STRIDE + kt * 32 + mh * 16];
      const v8bf alo = *(const v8bf*)pa;
      const v8bf ahi = *(const v8bf*)(pa + 8);
      const v16bf A = __builtin_shufflevector(alo, ahi,
          0, 1, 2, 3, 4, 5, 6, 7, 8, 9, 10, 11, 12, 13, 14, 15);
#pragma unroll
      for (int q = 0; q < 4; ++q)
#pragma unroll
        for (int j = 0; j < 2; ++j) {
          // B tile (32x16): lane n = nl, 16 contiguous K halves of row g.
          const int g = q * H + wv * 32 + j * 16 + nl;
          const v16bf Bm = *(const v16bf*)(wB + (size_t)g * KAUG + kt * 32 + mh * 16);
          acc[q][j] = __builtin_amdgcn_wmma_f32_16x16x32_bf16(
              false, A, false, Bm, (short)0, acc[q][j], false, false);
        }
    }

    // Element-wise LSTM cell on this wave's 16x32 slice (TRANS-pipe only:
    // v_exp/v_rcp/v_tanh — no IEEE divides on the serial critical path).
    v8f hnew[2];
#pragma unroll
    for (int j = 0; j < 2; ++j)
#pragma unroll
      for (int r = 0; r < 8; ++r) {
        const float iv = acc[0][j][r], fv = acc[1][j][r];
        const float gv = acc[2][j][r], ov = acc[3][j][r];
        const float si = fast_sigmoid(iv);
        const float sf = fast_sigmoid(fv);
        const float so = fast_sigmoid(ov);
        const float tg = fast_tanh(gv);
        const float c  = sf * cst[j][r] + si * tg;
        cst[j][r] = c;
        hnew[j][r] = so * fast_tanh(c);
      }

    __syncthreads();  // everyone is done reading hbuf for this step

#pragma unroll
    for (int j = 0; j < 2; ++j) {
      const int col = wv * 32 + j * 16 + nl;
#pragma unroll
      for (int r = 0; r < 8; ++r) {
        const int m = r + 8 * mh;
        const __bf16 hb = (__bf16)hnew[j][r];
        hbuf[m * LDS_STRIDE + col] = hb;
        hseq[(((size_t)lstm * 64 + bg * 16 + m) * T_STEPS + t) * H + col] = hb;
      }
    }
    if (t + 1 < T_STEPS && tid < 256) {
      const int m = tid >> 4, k = tid & 15;
      if (k < KIN) {
        float v = x[(((size_t)(bg * 16 + m)) * T_STEPS + (t + 1)) * 16 + XOFF + k];
        hbuf[m * LDS_STRIDE + H + k] = (__bf16)v;
      }
    }
    __syncthreads();  // h_{t+1} inputs ready
  }
}

// ---------------------------------------------------------------------------
// Kernel 4: output heads. 16 (b,t) rows per 256-thread block; 16 threads/row.
// cmd_out = h_cmd @ Wc^T + bc ; coord = tanh(( [h_coord|cmd_out] @ Wd^T + bd)*s)
// ---------------------------------------------------------------------------
__global__ void heads_kernel(const __bf16* __restrict__ hseq,
                             const float* __restrict__ wout_cmd,   // [10,512]
                             const float* __restrict__ bout_cmd,   // [10]
                             const float* __restrict__ wout_crd,   // [6,518]
                             const float* __restrict__ bout_crd,   // [6]
                             const float* __restrict__ scale_p,    // [1]
                             float* __restrict__ out_cmd,          // [65536,10]
                             float* __restrict__ out_crd) {        // [65536,6]
  __shared__ float cmdv[16][10];
  const int rloc = threadIdx.x >> 4;
  const int role = threadIdx.x & 15;
  const size_t row = (size_t)blockIdx.x * 16 + rloc;   // b*1024 + t
  const __bf16* hc = hseq + row * H;                   // cmd LSTM
  const __bf16* hd = hseq + (65536 + row) * H;         // coord LSTM

  if (role < 10) {
    float s = bout_cmd[role];
    const float* w = wout_cmd + (size_t)role * H;
    for (int k = 0; k < H; ++k) s += (float)hc[k] * w[k];
    cmdv[rloc][role] = s;
    out_cmd[row * 10 + role] = s;
  }
  __syncthreads();
  if (role < 6) {
    float s = bout_crd[role];
    const float* w = wout_crd + (size_t)role * 518;
    for (int k = 0; k < H; ++k) s += (float)hd[k] * w[k];
    for (int k = 0; k < 10; ++k) s += cmdv[rloc][k] * w[H + k];
    out_crd[row * 6 + role] = tanhf(s * scale_p[0]);
  }
}

// ---------------------------------------------------------------------------
extern "C" void kernel_launch(void* const* d_in, const int* in_sizes, int n_in,
                              void* d_out, int out_size, void* d_ws, size_t ws_size,
                              hipStream_t stream) {
  const float* x          = (const float*)d_in[0];
  const float* context    = (const float*)d_in[1];
  const float* w_cmd_emb  = (const float*)d_in[2];
  const float* b_cmd_emb  = (const float*)d_in[3];
  const float* w_crd_emb  = (const float*)d_in[4];
  const float* b_crd_emb  = (const float*)d_in[5];
  const float* w_ih_cmd   = (const float*)d_in[6];
  const float* w_hh_cmd   = (const float*)d_in[7];
  const float* b_ih_cmd   = (const float*)d_in[8];
  const float* b_hh_cmd   = (const float*)d_in[9];
  const float* w_ih_crd   = (const float*)d_in[10];
  const float* w_hh_crd   = (const float*)d_in[11];
  const float* b_ih_crd   = (const float*)d_in[12];
  const float* b_hh_crd   = (const float*)d_in[13];
  const float* w_out_cmd  = (const float*)d_in[14];
  const float* b_out_cmd  = (const float*)d_in[15];
  const float* w_out_crd  = (const float*)d_in[16];
  const float* b_out_crd  = (const float*)d_in[17];
  const float* scale_p    = (const float*)d_in[18];

  char* ws = (char*)d_ws;
  __bf16* whh_aug = (__bf16*)ws;                            // 4,456,448 B
  float*  ctxterm = (float*)(ws + 4456448);                 // 1,048,576 B
  __bf16* hseq    = (__bf16*)(ws + 4456448 + 1048576);      // 134,217,728 B

  float* out_cmd = (float*)d_out;                 // [64,1024,10]
  float* out_crd = out_cmd + (size_t)64 * 1024 * 10;  // [64,1024,6]

  prep_w_kernel<<<dim3(2048, 2), 256, 0, stream>>>(
      w_hh_cmd, w_ih_cmd, w_cmd_emb, w_hh_crd, w_ih_crd, w_crd_emb, whh_aug);

  prep_ctx_kernel<<<dim3(8, 64, 2), 256, 0, stream>>>(
      context, w_ih_cmd, b_cmd_emb, b_ih_cmd, b_hh_cmd,
      w_ih_crd, b_crd_emb, b_ih_crd, b_hh_crd, ctxterm);

  lstm_rec_kernel<<<dim3(4, 2), 512, 0, stream>>>(x, ctxterm, whh_aug, hseq);

  heads_kernel<<<dim3(4096), 256, 0, stream>>>(
      hseq, w_out_cmd, b_out_cmd, w_out_crd, b_out_crd, scale_p,
      out_cmd, out_crd);
}
